// SNN_Model_80857054314830
// MI455X (gfx1250) — matrix operand order repaired and checked by reading, not compiled, not run on hardware
//
#include <hip/hip_runtime.h>
#include <hip/hip_bf16.h>
#include <stdint.h>

typedef __attribute__((ext_vector_type(16))) _Float16 v16h;
typedef __attribute__((ext_vector_type(8)))  _Float16 v8h;
typedef __attribute__((ext_vector_type(8)))  float    v8f;

#define B_      64
#define T_      100

// ---------------------------------------------------------------------------
__global__ void k_zero(float* __restrict__ p, int n) {
  int i = blockIdx.x * blockDim.x + threadIdx.x;
  if (i < n) p[i] = 0.f;
}

// ---------------------------------------------------------------------------
// Fragment-ready B swizzle: W [O][Kd] f32 -> Bs f16 blocks.
// Block (kc, nt) is 1KB: lane l owns 16 contiguous halves = B[kc*32 + (l>>4)*16 + j][nt*16 + (l&15)]
__global__ void k_cvt_bs(const float* __restrict__ W, _Float16* __restrict__ Bs,
                         int O, int Kd, int NT) {
  int idx = blockIdx.x * blockDim.x + threadIdx.x;
  int total = (Kd >> 5) * NT * 512;
  if (idx >= total) return;
  int j    = idx & 15;
  int lane = (idx >> 4) & 31;
  int t    = idx >> 9;                 // kc*NT + nt
  int nt = t % NT, kc = t / NT;
  int n  = (nt << 4) + (lane & 15);
  int Kk = (kc << 5) + ((lane >> 4) << 4) + j;
  Bs[idx] = (n < O) ? (_Float16)W[(size_t)n * Kd + Kk] : (_Float16)0.f;
}

// ---------------------------------------------------------------------------
__global__ void k_pois(const float* __restrict__ r, const float* __restrict__ x,
                       float* __restrict__ p0, int n) {
  int i = blockIdx.x * blockDim.x + threadIdx.x;
  if (i >= n) return;
  float xx = x[i];
  float ax = fabsf(xx);
  float sg = (xx > 0.f) ? 1.f : ((xx < 0.f) ? -1.f : 0.f);
  p0[i] = (r[i] <= ax) ? sg : 0.f;
}

// ---------------------------------------------------------------------------
__global__ void k_layer1(const float* __restrict__ p0, const float* __restrict__ Wc1,
                         float* __restrict__ m0, _Float16* __restrict__ p1) {
  __shared__ float w[288];
  int tid = threadIdx.x;
  for (int i = tid; i < 288; i += 256) w[i] = Wc1[i];
  __syncthreads();
  int idx = blockIdx.x * 256 + tid;
  if (idx >= B_ * 32 * 14 * 14) return;
  int px = idx % 14; int t = idx / 14;
  int py = t % 14;   t /= 14;
  int co = t % 32;   int b = t / 32;
  const float* wp  = w + co * 9;
  const float* img = p0 + b * 784;
  float* mrow = m0 + (size_t)(b * 32 + co) * 784;
  float ssum = 0.f;
#pragma unroll
  for (int dy = 0; dy < 2; ++dy)
#pragma unroll
  for (int dx = 0; dx < 2; ++dx) {
    int y = 2 * py + dy, x = 2 * px + dx;
    int mi = y * 28 + x;
    float mm = mrow[mi];
    float sp = (mm > 1.f) ? 1.f : 0.f;
    ssum += sp;
    float conv = 0.f;
#pragma unroll
    for (int ky = 0; ky < 3; ++ky)
#pragma unroll
    for (int kx = 0; kx < 3; ++kx) {
      int iy = y + ky - 1, ix = x + kx - 1;
      if ((unsigned)iy < 28u && (unsigned)ix < 28u)
        conv += img[iy * 28 + ix] * wp[ky * 3 + kx];
    }
    mrow[mi] = mm + conv - sp;
  }
  p1[idx] = (_Float16)(0.25f * ssum);
}

// ---------------------------------------------------------------------------
__global__ void k_spike2(float* __restrict__ m2, _Float16* __restrict__ p2) {
  int idx = blockIdx.x * blockDim.x + threadIdx.x;
  if (idx >= B_ * 64 * 49) return;
  int px = idx % 7; int t = idx / 7;
  int py = t % 7;   t /= 7;
  int ci = t % 64;  int b = t / 64;
  float* mrow = m2 + (size_t)(b * 64 + ci) * 196;
  float sum = 0.f;
#pragma unroll
  for (int dy = 0; dy < 2; ++dy)
#pragma unroll
  for (int dx = 0; dx < 2; ++dx) {
    int mi = (2 * py + dy) * 14 + 2 * px + dx;
    float mm = mrow[mi];
    float sp = (mm > 1.f) ? 1.f : 0.f;
    sum += sp;
    mrow[mi] = mm - sp;
  }
  p2[b * 3136 + ci * 49 + py * 7 + px] = (_Float16)(0.25f * sum);
}

// ---------------------------------------------------------------------------
__global__ void k_spike_fc(float* __restrict__ m, const float* __restrict__ mask,
                           _Float16* __restrict__ s, int n) {
  int i = blockIdx.x * blockDim.x + threadIdx.x;
  if (i >= n) return;
  float mm = m[i];
  float sp = (mm > 1.f) ? 1.f : 0.f;
  m[i] = mm - sp;
  s[i] = (_Float16)(sp * mask[i]);
}

// ---------------------------------------------------------------------------
// WMMA GEMM with accumulate: C[64][ldc] += A[64][K] * Bswz
// Templated thread count -> straight-line staging via async LDS loads.
template<int THREADS>
__global__ void k_gemm_wmma(const _Float16* __restrict__ A,
                            const _Float16* __restrict__ Bs,
                            float* __restrict__ C,
                            int K, int NT, int ldc, int nlimit) {
  __shared__ _Float16 As[64 * 32];
  const int tid  = threadIdx.x;
  const int wave = tid >> 5, lane = tid & 31;
  const int hf = lane >> 4, m = lane & 15;
  const int mi = wave & 3, nj = wave >> 2;
  constexpr int TPB = THREADS >> 7;                  // n-groups per block
  const int ntile0 = (blockIdx.x * TPB + nj) << 1;   // first of 2 n-tiles
  const int kchunks = K >> 5;
  const int bstride = NT << 9;                       // halves per k-chunk
  const _Float16* bbase = Bs + ((ntile0 << 5) + lane) * 16;
  const _Float16* arow  = As + ((mi << 4) + m) * 32 + (hf << 4);
  // staging: 64x32 chunk = 256 x b128; LOADS per thread
  constexpr int LOADS = 256 / THREADS;
  uint32_t        ldsoff[LOADS];
  const _Float16* gsrc[LOADS];
#pragma unroll
  for (int u = 0; u < LOADS; ++u) {
    int i = tid + u * THREADS;
    ldsoff[u] = (uint32_t)(size_t)(As + (i >> 2) * 32 + ((i & 3) << 3));
    gsrc[u]   = A + (i >> 2) * K + ((i & 3) << 3);
  }
  v8f acc0 = {0.f, 0.f, 0.f, 0.f, 0.f, 0.f, 0.f, 0.f};
  v8f acc1 = acc0;
  for (int kc = 0; kc < kchunks; ++kc) {
    const int k0 = kc << 5;
#pragma unroll
    for (int u = 0; u < LOADS; ++u) {                 // async gmem->LDS (ASYNCcnt)
      uint64_t ga = (uint64_t)(size_t)(gsrc[u] + k0);
      asm volatile("global_load_async_to_lds_b128 %0, %1, off"
                   :: "v"(ldsoff[u]), "v"(ga) : "memory");
    }
    asm volatile("s_wait_asynccnt 0x0" ::: "memory");
    __syncthreads();
    v8h alo = *(const v8h*)(arow);
    v8h ahi = *(const v8h*)(arow + 8);
    v16h af;
#pragma unroll
    for (int i = 0; i < 8; ++i) { af[i] = alo[i]; af[i + 8] = ahi[i]; }
    const _Float16* bp = bbase + kc * bstride;
    __builtin_prefetch(bp + bstride, 0, 1);           // global_prefetch_b8
    v16h bf0 = *(const v16h*)(bp);
    v16h bf1 = *(const v16h*)(bp + 512);
    acc0 = __builtin_amdgcn_wmma_f32_16x16x32_f16(false, af, false, bf0,
                                                  (short)0, acc0, false, false);
    acc1 = __builtin_amdgcn_wmma_f32_16x16x32_f16(false, af, false, bf1,
                                                  (short)0, acc1, false, false);
    __syncthreads();
  }
  const int n0   = (ntile0 << 4) + m;
  const int row0 = (mi << 4) + (hf << 3);
  if (n0 < nlimit) {
#pragma unroll
    for (int r = 0; r < 8; ++r) C[(row0 + r) * ldc + n0] += acc0[r];
  }
  if (n0 + 16 < nlimit) {
#pragma unroll
    for (int r = 0; r < 8; ++r) C[(row0 + r) * ldc + n0 + 16] += acc1[r];
  }
}

// ---------------------------------------------------------------------------
// conv2 implicit GEMM via WMMA: 64 rows x 64 cols per block, K=288 (Bs NT=4).
__global__ void k_conv2_wmma(const _Float16* __restrict__ p1,
                             const _Float16* __restrict__ Bs,
                             float* __restrict__ m2) {
  __shared__ _Float16 As[64 * 304];                   // 38912 B
  const int tid  = threadIdx.x;                       // 256
  const int wave = tid >> 5, lane = tid & 31;
  const int hf = lane >> 4, m = lane & 15;
  const int mi = wave & 3, nj = wave >> 2;            // 4 M-tiles x 2 n-groups
  const int rowbase = blockIdx.x << 6;
  {                                                   // im2col gather: thread = (row, 8-ci strip)
    const int R   = tid >> 2;                         // 0..63
    const int ciq = (tid & 3) << 3;                   // 0,8,16,24
    int r = rowbase + R;
    int b = r / 196, rem = r - b * 196;
    int y = rem / 14, xx = rem - y * 14;
    _Float16* dst = As + R * 304;
    const _Float16* src = p1 + (b * 32 + ciq) * 196;
    for (int ci = 0; ci < 8; ++ci) {
#pragma unroll
      for (int ky = 0; ky < 3; ++ky)
#pragma unroll
      for (int kx = 0; kx < 3; ++kx) {
        int iy = y + ky - 1, ix = xx + kx - 1;
        _Float16 v = (_Float16)0.f;
        if ((unsigned)iy < 14u && (unsigned)ix < 14u)
          v = src[ci * 196 + iy * 14 + ix];
        dst[(ciq + ci) * 9 + ky * 3 + kx] = v;
      }
    }
  }
  __syncthreads();
  const _Float16* arow  = As + ((mi << 4) + m) * 304 + (hf << 4);
  const int ntile0 = nj << 1;
  const _Float16* bbase = Bs + ((ntile0 << 5) + lane) * 16;
  v8f acc0 = {0.f, 0.f, 0.f, 0.f, 0.f, 0.f, 0.f, 0.f};
  v8f acc1 = acc0;
#pragma unroll
  for (int kc = 0; kc < 9; ++kc) {
    const int k0 = kc << 5;
    v8h alo = *(const v8h*)(arow + k0);
    v8h ahi = *(const v8h*)(arow + k0 + 8);
    v16h af;
#pragma unroll
    for (int i = 0; i < 8; ++i) { af[i] = alo[i]; af[i + 8] = ahi[i]; }
    const _Float16* bp = bbase + kc * 2048;           // NT=4
    v16h bf0 = *(const v16h*)(bp);
    v16h bf1 = *(const v16h*)(bp + 512);
    acc0 = __builtin_amdgcn_wmma_f32_16x16x32_f16(false, af, false, bf0,
                                                  (short)0, acc0, false, false);
    acc1 = __builtin_amdgcn_wmma_f32_16x16x32_f16(false, af, false, bf1,
                                                  (short)0, acc1, false, false);
  }
  const int n0 = (ntile0 << 4) + m;
#pragma unroll
  for (int r = 0; r < 8; ++r) {
    int grow = rowbase + (mi << 4) + (hf << 3) + r;
    int b = grow / 196, rem = grow - b * 196;
    float* mrow = m2 + (size_t)b * 12544 + rem;       // [b][co][rem], co stride 196
    mrow[n0 * 196]        += acc0[r];
    mrow[(n0 + 16) * 196] += acc1[r];
  }
}

// ---------------------------------------------------------------------------
extern "C" void kernel_launch(void* const* d_in, const int* in_sizes, int n_in,
                              void* d_out, int out_size, void* d_ws, size_t ws_size,
                              hipStream_t stream) {
  (void)in_sizes; (void)n_in; (void)out_size; (void)ws_size;
  const float* x      = (const float*)d_in[0];
  const float* rand_t = (const float*)d_in[1];
  const float* Wc1    = (const float*)d_in[2];
  const float* Wc2    = (const float*)d_in[3];
  const float* W1     = (const float*)d_in[4];
  const float* W2     = (const float*)d_in[5];
  const float* W3     = (const float*)d_in[6];
  const float* mask1  = (const float*)d_in[7];
  const float* mask2  = (const float*)d_in[8];
  float* out = (float*)d_out;

  char*  ws  = (char*)d_ws;
  size_t off = 0;
  auto carve = [&](size_t bytes) -> char* {
    char* p = ws + off;
    off = (off + bytes + 255) & ~(size_t)255;
    return p;
  };
  float*    m0   = (float*)   carve(1605632u * 4);  // [64,32,28,28]
  float*    m2   = (float*)   carve( 802816u * 4);  // [64,64,14,14]
  float*    m5   = (float*)   carve(  32768u * 4);  // [64,512]
  float*    m8   = (float*)   carve(  32768u * 4);  // [64,512]
  float*    p0   = (float*)   carve(  50176u * 4);  // [64,28,28]
  _Float16* p1   = (_Float16*)carve( 401408u * 2);  // [64,32,14,14]
  _Float16* p2   = (_Float16*)carve( 200704u * 2);  // [64,3136]
  _Float16* s5m  = (_Float16*)carve(  32768u * 2);  // [64,512]
  _Float16* s8m  = (_Float16*)carve(  32768u * 2);  // [64,512]
  _Float16* W1s  = (_Float16*)carve(1605632u * 2);  // swz K=3136 NT=32
  _Float16* W2s  = (_Float16*)carve( 262144u * 2);  // swz K=512  NT=32
  _Float16* W3s  = (_Float16*)carve(  16384u * 2);  // swz K=512  NT=2 (pad O=10->32)
  _Float16* Wc2s = (_Float16*)carve(  18432u * 2);  // swz K=288  NT=4

  k_zero<<<6272, 256, 0, stream>>>(m0, 1605632);
  k_zero<<<3136, 256, 0, stream>>>(m2,  802816);
  k_zero<<< 128, 256, 0, stream>>>(m5,   32768);
  k_zero<<< 128, 256, 0, stream>>>(m8,   32768);
  k_zero<<<   3, 256, 0, stream>>>(out,    640);

  k_cvt_bs<<<6272, 256, 0, stream>>>(W1,  W1s,  512, 3136, 32);
  k_cvt_bs<<<1024, 256, 0, stream>>>(W2,  W2s,  512,  512, 32);
  k_cvt_bs<<<  64, 256, 0, stream>>>(W3,  W3s,   10,  512,  2);
  k_cvt_bs<<<  72, 256, 0, stream>>>(Wc2, Wc2s,  64,  288,  4);

  for (int t = 0; t < T_; ++t) {
    k_pois        <<<196, 256, 0, stream>>>(rand_t + (size_t)t * 50176, x, p0, 50176);
    k_layer1      <<<1568, 256, 0, stream>>>(p0, Wc1, m0, p1);
    k_spike2      <<<784, 256, 0, stream>>>(m2, p2);
    k_conv2_wmma  <<<196, 256, 0, stream>>>(p1, Wc2s, m2);
    k_spike_fc    <<<128, 256, 0, stream>>>(m5, mask1, s5m, 32768);
    k_gemm_wmma<256><<<8, 256, 0, stream>>>(p2,  W1s, m5, 3136, 32, 512, 512);
    k_spike_fc    <<<128, 256, 0, stream>>>(m8, mask2, s8m, 32768);
    k_gemm_wmma<256><<<8, 256, 0, stream>>>(s5m, W2s, m8,  512, 32, 512, 512);
    k_gemm_wmma<128><<<1, 128, 0, stream>>>(s8m, W3s, out, 512,  2,  10,  10);
  }
}
